// AsymmetricAttention_43731357007980
// MI455X (gfx1250) — matrix operand order repaired and checked by reading, not compiled, not run on hardware
//
#include <hip/hip_runtime.h>
#include <math.h>

#define N_   2048
#define L_   256
#define T_   (N_ + L_)
#define DX_  1536
#define DY_  768
#define H_   12
#define D_   128
#define QKV_ (3 * DX_) /* 4608 */
#define EPS_MOD 1e-6f
#define EPS_QK  1e-5f

typedef __attribute__((ext_vector_type(16))) _Float16 v16h;
typedef __attribute__((ext_vector_type(8)))  _Float16 v8h;
typedef __attribute__((ext_vector_type(8)))  float    v8f;
typedef __attribute__((ext_vector_type(4)))  int      i32x4;

// ---------------------------------------------------------------------------
// CDNA5 async global->LDS copy support (gfx1250, ASYNCcnt-tracked)
// Builtin signature (from toolchain diagnostics): params are v4i* in
// address_space(1) (global src) and address_space(3) (LDS dst), + 2 imms.
// ---------------------------------------------------------------------------
#if __has_builtin(__builtin_amdgcn_global_load_async_to_lds_b128)
#define USE_ASYNC_LDS 1
#else
#define USE_ASYNC_LDS 0
#endif

#if __has_builtin(__builtin_amdgcn_s_wait_asynccnt)
#define WAIT_ASYNC(n) __builtin_amdgcn_s_wait_asynccnt(n)
#else
#define WAIT_ASYNC(n) asm volatile("s_wait_asynccnt %0" ::"i"(n) : "memory")
#endif

#if USE_ASYNC_LDS
typedef __attribute__((address_space(1))) i32x4 gl_i32x4;
typedef __attribute__((address_space(3))) i32x4 ld_i32x4;

static __device__ __forceinline__ void async_copy16(const _Float16* g, _Float16* l) {
  __builtin_amdgcn_global_load_async_to_lds_b128((gl_i32x4*)g, (ld_i32x4*)l, 0, 0);
}
#endif

// ---------------------------------------------------------------------------
// WMMA fragment loaders (CDNA5 16-bit layouts, wave32; lane = lo + 16*hi)
// A (16xK tile, M=row):  halves [0..7]=K(k0+hi*8 ..), [8..15]=K(k0+16+hi*8 ..)
// B (Kx16 tile, via Bt[n][k] row-major): halves [0..7]=K(k0+hi*8 ..), [8..15]=K(k0+16+hi*8 ..)
// ---------------------------------------------------------------------------
static __device__ __forceinline__ v16h load_frag_a(const _Float16* base, int ld,
                                                   int row0, int k0, int lane) {
  const int lo = lane & 15, hi = lane >> 4;
  const _Float16* p = base + (size_t)(row0 + lo) * ld + k0 + hi * 8;
  v8h x0 = *(const v8h*)p;
  v8h x1 = *(const v8h*)(p + 16);
  v16h r;
#pragma unroll
  for (int i = 0; i < 8; ++i) { r[i] = x0[i]; r[8 + i] = x1[i]; }
  return r;
}

static __device__ __forceinline__ v16h load_frag_b(const _Float16* bt, int ld,
                                                   int n0, int k0, int lane) {
  const int lo = lane & 15, hi = lane >> 4;
  const _Float16* p = bt + (size_t)(n0 + lo) * ld + k0 + hi * 8;
  v8h x0 = *(const v8h*)p;
  v8h x1 = *(const v8h*)(p + 16);
  v16h r;
#pragma unroll
  for (int i = 0; i < 8; ++i) { r[i] = x0[i]; r[8 + i] = x1[i]; }
  return r;
}

static __device__ __forceinline__ v8f wmma_f16(v16h a, v16h b, v8f c) {
  return __builtin_amdgcn_wmma_f32_16x16x32_f16(false, a, false, b, (short)0, c,
                                                false, false);
}

// ---------------------------------------------------------------------------
// 1) RMS-norm + modulation: out = f16( x * rsqrt(mean(x^2)+eps) * (1+scale) )
// ---------------------------------------------------------------------------
__global__ __launch_bounds__(256)
void rmsmod_kernel(const float* __restrict__ X, const float* __restrict__ scale,
                   _Float16* __restrict__ out, int Drow) {
  const int row = blockIdx.x;
  const int tid = threadIdx.x;
  const int lane = tid & 31, wid = tid >> 5;
  const float* xr = X + (size_t)row * Drow;
  float s = 0.f;
  for (int i = tid; i < Drow; i += 256) { float v = xr[i]; s += v * v; }
#pragma unroll
  for (int m = 16; m >= 1; m >>= 1) s += __shfl_xor(s, m, 32);
  __shared__ float red[8];
  if (lane == 0) red[wid] = s;
  __syncthreads();
  float tot = 0.f;
#pragma unroll
  for (int i = 0; i < 8; ++i) tot += red[i];
  const float rinv = rsqrtf(tot / (float)Drow + EPS_MOD);
  _Float16* orow = out + (size_t)row * Drow;
  for (int i = tid; i < Drow; i += 256)
    orow[i] = (_Float16)(xr[i] * rinv * (1.f + scale[i]));
}

// ---------------------------------------------------------------------------
// 2) Weight transpose + f32 -> f16:  Wt[n][k] = W[k][n]
// ---------------------------------------------------------------------------
__global__ __launch_bounds__(256)
void transpose_f16_kernel(const float* __restrict__ W, _Float16* __restrict__ Wt,
                          int K, int Nout) {
  __shared__ float tile[32][33];
  const int nb = blockIdx.x * 32, kb = blockIdx.y * 32;
  const int tx = threadIdx.x & 31, ty = threadIdx.x >> 5; // 32 x 8
#pragma unroll
  for (int i = 0; i < 32; i += 8) {
    int k = kb + ty + i, n = nb + tx;
    tile[ty + i][tx] = (k < K && n < Nout) ? W[(size_t)k * Nout + n] : 0.f;
  }
  __syncthreads();
#pragma unroll
  for (int i = 0; i < 32; i += 8) {
    int n = nb + ty + i, k = kb + tx;
    if (n < Nout && k < K) Wt[(size_t)n * K + k] = (_Float16)tile[tx][ty + i];
  }
}

// ---------------------------------------------------------------------------
// 3) WMMA GEMM: C[M x Nout] = A[M x K](f16) * Wt[Nout x K]^T(f16) + bias (f32)
// block = 8 waves -> 128(M) x 64(N); wave -> 32x32 (2x2 WMMA tiles).
// Async path: A(128x32) and B(64x32) tiles staged in LDS via
// global_load_async_to_lds_b128, double-buffered (DMA of tile k+1 overlaps
// WMMAs on tile k); fragments then come from ds_load_b128.
// M % 128 == 0, Nout % 64 == 0, K % 32 == 0 for every call in this pipeline.
// ---------------------------------------------------------------------------
__global__ __launch_bounds__(256)
void gemm_wmma_f16(const _Float16* __restrict__ A, const _Float16* __restrict__ Wt,
                   const float* __restrict__ bias, float* __restrict__ C,
                   int M, int K, int Nout) {
  const int tid = threadIdx.x;
  const int lane = tid & 31;
  const int w = tid >> 5;
  const int mw = w >> 1, nw = w & 1;
  const int rowB = blockIdx.x * 128;
  const int colB = blockIdx.y * 64;
  const int lo = lane & 15, hi = lane >> 4;

  v8f acc[2][2] = {};

#if USE_ASYNC_LDS
  __shared__ _Float16 lA[2][128][48]; // 96B row stride: 16B aligned, low conflicts
  __shared__ _Float16 lB[2][64][48];
  const int ra = tid >> 2, qa = tid & 3; // A: thread -> rows ra and ra+64, 16B chunk qa
  const int rb = tid >> 2, qb = tid & 3; // B: rows 0..63

  const int nk = K >> 5;
#define GEMM_ISSUE(buf_, k0_)                                                   \
  do {                                                                          \
    async_copy16(A + (size_t)(rowB + ra) * K + (k0_) + qa * 8,                  \
                 &lA[(buf_)][ra][qa * 8]);                                      \
    async_copy16(A + (size_t)(rowB + 64 + ra) * K + (k0_) + qa * 8,             \
                 &lA[(buf_)][64 + ra][qa * 8]);                                 \
    async_copy16(Wt + (size_t)(colB + rb) * K + (k0_) + qb * 8,                 \
                 &lB[(buf_)][rb][qb * 8]);                                      \
  } while (0)

  GEMM_ISSUE(0, 0);
  WAIT_ASYNC(0);
  __syncthreads();
  for (int kt = 0; kt < nk; ++kt) {
    const int buf = kt & 1;
    if (kt + 1 < nk) GEMM_ISSUE(buf ^ 1, (kt + 1) << 5); // overlap DMA with WMMA
    v16h a0 = load_frag_a(&lA[buf][0][0], 48, mw * 32, 0, lane);
    v16h a1 = load_frag_a(&lA[buf][0][0], 48, mw * 32 + 16, 0, lane);
    v16h b0 = load_frag_b(&lB[buf][0][0], 48, nw * 32, 0, lane);
    v16h b1 = load_frag_b(&lB[buf][0][0], 48, nw * 32 + 16, 0, lane);
    acc[0][0] = wmma_f16(a0, b0, acc[0][0]);
    acc[0][1] = wmma_f16(a0, b1, acc[0][1]);
    acc[1][0] = wmma_f16(a1, b0, acc[1][0]);
    acc[1][1] = wmma_f16(a1, b1, acc[1][1]);
    if (kt + 1 < nk) {
      WAIT_ASYNC(0);
      __syncthreads();
    }
  }
#undef GEMM_ISSUE
#else
  for (int k0 = 0; k0 < K; k0 += 32) {
    if (k0 + 32 < K) {
      __builtin_prefetch(A + (size_t)(rowB + mw * 32 + lo) * K + k0 + 32, 0, 1);
      __builtin_prefetch(Wt + (size_t)(colB + nw * 32 + lo) * K + k0 + 32, 0, 1);
    }
    v16h a0 = load_frag_a(A, K, rowB + mw * 32, k0, lane);
    v16h a1 = load_frag_a(A, K, rowB + mw * 32 + 16, k0, lane);
    v16h b0 = load_frag_b(Wt, K, colB + nw * 32, k0, lane);
    v16h b1 = load_frag_b(Wt, K, colB + nw * 32 + 16, k0, lane);
    acc[0][0] = wmma_f16(a0, b0, acc[0][0]);
    acc[0][1] = wmma_f16(a0, b1, acc[0][1]);
    acc[1][0] = wmma_f16(a1, b0, acc[1][0]);
    acc[1][1] = wmma_f16(a1, b1, acc[1][1]);
  }
#endif

#pragma unroll
  for (int mi = 0; mi < 2; ++mi) {
#pragma unroll
    for (int ni = 0; ni < 2; ++ni) {
      const int col = colB + nw * 32 + ni * 16 + lo;
      const float bv = bias[col];
#pragma unroll
      for (int r = 0; r < 8; ++r) {
        const int row = rowB + mw * 32 + mi * 16 + r + 8 * hi;
        C[(size_t)row * Nout + col] = acc[mi][ni][r] + bv;
      }
    }
  }
}

// ---------------------------------------------------------------------------
// 4) Per-head QK RMS-norm (+RoPE for x tokens); builds Q,K [H][T][D] f16 and
//    V^T [H][D][T] f16. One 128-thread block per (token, head).
// ---------------------------------------------------------------------------
__global__ __launch_bounds__(128)
void qknorm_rope_kernel(const float* __restrict__ qkv,
                        const float* __restrict__ rope_cos,
                        const float* __restrict__ rope_sin,
                        const float* __restrict__ qnx, const float* __restrict__ knx,
                        const float* __restrict__ qny, const float* __restrict__ kny,
                        _Float16* __restrict__ Qb, _Float16* __restrict__ Kb,
                        _Float16* __restrict__ Vt) {
  const int t = blockIdx.x / H_;
  const int h = blockIdx.x % H_;
  const int d = threadIdx.x;
  const int lane = threadIdx.x & 31, wid = threadIdx.x >> 5;
  const bool is_x = (t < N_);
  const float* row = qkv + (size_t)t * QKV_;
  const float q = row[0 * DX_ + h * D_ + d];
  const float k = row[1 * DX_ + h * D_ + d];
  const float v = row[2 * DX_ + h * D_ + d];

  __shared__ float redq[4], redk[4];
  float sq = q * q, sk = k * k;
#pragma unroll
  for (int m = 16; m >= 1; m >>= 1) {
    sq += __shfl_xor(sq, m, 32);
    sk += __shfl_xor(sk, m, 32);
  }
  if (lane == 0) { redq[wid] = sq; redk[wid] = sk; }
  __syncthreads();
  const float qs = redq[0] + redq[1] + redq[2] + redq[3];
  const float ks = redk[0] + redk[1] + redk[2] + redk[3];
  const float qr = rsqrtf(qs * (1.f / D_) + EPS_QK);
  const float kr = rsqrtf(ks * (1.f / D_) + EPS_QK);
  float qn = q * qr * (is_x ? qnx[d] : qny[d]);
  float kn = k * kr * (is_x ? knx[d] : kny[d]);

  if (is_x) {
    const size_t ci = ((size_t)t * H_ + h) * (D_ / 2) + (d >> 1);
    const float c = rope_cos[ci], s = rope_sin[ci];
    const float qp = __shfl_xor(qn, 1, 32); // even<->odd pair (same wave)
    const float kp = __shfl_xor(kn, 1, 32);
    const bool even = (d & 1) == 0;
    qn = even ? (qn * c - qp * s) : (qp * s + qn * c);
    kn = even ? (kn * c - kp * s) : (kp * s + kn * c);
  }
  const size_t qi = ((size_t)h * T_ + t) * D_ + d;
  Qb[qi] = (_Float16)qn;
  Kb[qi] = (_Float16)kn;
  Vt[((size_t)h * D_ + d) * T_ + t] = (_Float16)v;
}

// ---------------------------------------------------------------------------
// 5) Flash attention with WMMA. grid = (T/128, H), block = 8 waves.
// Each wave: 16 query rows, streams 32-key tiles; S=QK^T (8 wmma),
// online softmax, P via LDS relayout, O += P*V (8 wmma).
// ---------------------------------------------------------------------------
__global__ __launch_bounds__(256)
void attn_wmma_kernel(const _Float16* __restrict__ Qb, const _Float16* __restrict__ Kb,
                      const _Float16* __restrict__ Vt, _Float16* __restrict__ Out) {
  __shared__ _Float16 pbuf[8][16][48]; // per-wave 16x32 P tile, 96B row stride
  const int lane = threadIdx.x & 31;
  const int w = threadIdx.x >> 5;
  const int h = blockIdx.y;
  const int q0 = blockIdx.x * 128 + w * 16;
  const int lo = lane & 15, hi = lane >> 4;

  const _Float16* Qh = Qb + (size_t)h * T_ * D_;
  const _Float16* Kh = Kb + (size_t)h * T_ * D_;
  const _Float16* Vh = Vt + (size_t)h * D_ * T_;

  v16h qf[4];
#pragma unroll
  for (int kd = 0; kd < 4; ++kd) qf[kd] = load_frag_a(Qh, D_, q0, kd * 32, lane);

  v8f o[8] = {};
  float mrow[8], lrow[8];
#pragma unroll
  for (int r = 0; r < 8; ++r) { mrow[r] = -3.0e38f; lrow[r] = 0.f; }
  const float scl = 0.08838834764831845f; // 1/sqrt(128)

  for (int key0 = 0; key0 < T_; key0 += 32) {
    v8f s0 = {}, s1 = {};
#pragma unroll
    for (int kd = 0; kd < 4; ++kd) {
      v16h b0 = load_frag_b(Kh, D_, key0, kd * 32, lane);
      v16h b1 = load_frag_b(Kh, D_, key0 + 16, kd * 32, lane);
      s0 = wmma_f16(qf[kd], b0, s0);
      s1 = wmma_f16(qf[kd], b1, s1);
    }
    float p0[8], p1[8], alpha[8];
#pragma unroll
    for (int r = 0; r < 8; ++r) {
      const float a = s0[r] * scl, b = s1[r] * scl;
      float mx = fmaxf(a, b);
#pragma unroll
      for (int msk = 8; msk >= 1; msk >>= 1) mx = fmaxf(mx, __shfl_xor(mx, msk, 32));
      const float mn = fmaxf(mrow[r], mx);
      alpha[r] = __expf(mrow[r] - mn);
      mrow[r] = mn;
      p0[r] = __expf(a - mn);
      p1[r] = __expf(b - mn);
      float ps = p0[r] + p1[r];
#pragma unroll
      for (int msk = 8; msk >= 1; msk >>= 1) ps += __shfl_xor(ps, msk, 32);
      lrow[r] = lrow[r] * alpha[r] + ps;
    }
#pragma unroll
    for (int i = 0; i < 8; ++i)
#pragma unroll
      for (int r = 0; r < 8; ++r) o[i][r] *= alpha[r];

    // C-layout -> A-layout through wave-private LDS (LDS same-wave in-order)
#pragma unroll
    for (int r = 0; r < 8; ++r) {
      pbuf[w][r + 8 * hi][lo]      = (_Float16)p0[r];
      pbuf[w][r + 8 * hi][16 + lo] = (_Float16)p1[r];
    }
    v16h pf = load_frag_a(&pbuf[w][0][0], 48, 0, 0, lane);
#pragma unroll
    for (int dt = 0; dt < 8; ++dt) {
      v16h bv = load_frag_b(Vh, T_, dt * 16, key0, lane);
      o[dt] = wmma_f16(pf, bv, o[dt]);
    }
  }
#pragma unroll
  for (int dt = 0; dt < 8; ++dt)
#pragma unroll
    for (int r = 0; r < 8; ++r) {
      const int tok = q0 + r + 8 * hi;
      Out[(size_t)tok * (H_ * D_) + h * D_ + dt * 16 + lo] =
          (_Float16)(o[dt][r] / lrow[r]);
    }
}

// ---------------------------------------------------------------------------
extern "C" void kernel_launch(void* const* d_in, const int* in_sizes, int n_in,
                              void* d_out, int out_size, void* d_ws, size_t ws_size,
                              hipStream_t stream) {
  const float* x        = (const float*)d_in[0];
  const float* y        = (const float*)d_in[1];
  const float* scale_x  = (const float*)d_in[2];
  const float* scale_y  = (const float*)d_in[3];
  const float* rope_cos = (const float*)d_in[4];
  const float* rope_sin = (const float*)d_in[5];
  const float* Wqkv_x   = (const float*)d_in[6];
  const float* bqkv_x   = (const float*)d_in[7];
  const float* Wqkv_y   = (const float*)d_in[8];
  const float* bqkv_y   = (const float*)d_in[9];
  const float* qnx      = (const float*)d_in[10];
  const float* knx      = (const float*)d_in[11];
  const float* qny      = (const float*)d_in[12];
  const float* kny      = (const float*)d_in[13];
  const float* Wproj_x  = (const float*)d_in[14];
  const float* bproj_x  = (const float*)d_in[15];
  const float* Wproj_y  = (const float*)d_in[16];
  const float* bproj_y  = (const float*)d_in[17];

  char* ws = (char*)d_ws;
  size_t off = 0;
  auto alloc = [&](size_t bytes) -> void* {
    void* p = ws + off;
    off = (off + bytes + 255) & ~(size_t)255;
    return p;
  };
  _Float16* xm    = (_Float16*)alloc((size_t)N_ * DX_ * 2);
  _Float16* ym    = (_Float16*)alloc((size_t)L_ * DY_ * 2);
  _Float16* WtQX  = (_Float16*)alloc((size_t)QKV_ * DX_ * 2);
  _Float16* WtQY  = (_Float16*)alloc((size_t)QKV_ * DY_ * 2);
  _Float16* WtPX  = (_Float16*)alloc((size_t)DX_ * DX_ * 2);
  _Float16* WtPY  = (_Float16*)alloc((size_t)DY_ * DX_ * 2);
  float*    qkv   = (float*)alloc((size_t)T_ * QKV_ * 4);
  _Float16* Qb    = (_Float16*)alloc((size_t)H_ * T_ * D_ * 2);
  _Float16* Kb    = (_Float16*)alloc((size_t)H_ * T_ * D_ * 2);
  _Float16* Vt    = (_Float16*)alloc((size_t)H_ * T_ * D_ * 2);
  _Float16* aout  = (_Float16*)alloc((size_t)T_ * DX_ * 2);

  // 1) modulated RMS norm -> f16 activations
  rmsmod_kernel<<<N_, 256, 0, stream>>>(x, scale_x, xm, DX_);
  rmsmod_kernel<<<L_, 256, 0, stream>>>(y, scale_y, ym, DY_);

  // 2) weight transposes (f32 -> f16, K-contiguous per output column)
  transpose_f16_kernel<<<dim3(QKV_ / 32, DX_ / 32), 256, 0, stream>>>(Wqkv_x, WtQX, DX_, QKV_);
  transpose_f16_kernel<<<dim3(QKV_ / 32, DY_ / 32), 256, 0, stream>>>(Wqkv_y, WtQY, DY_, QKV_);
  transpose_f16_kernel<<<dim3(DX_ / 32, DX_ / 32), 256, 0, stream>>>(Wproj_x, WtPX, DX_, DX_);
  transpose_f16_kernel<<<dim3(DY_ / 32, DX_ / 32), 256, 0, stream>>>(Wproj_y, WtPY, DX_, DY_);

  // 3) QKV GEMMs (rows 0..N-1 from x, rows N..T-1 from y)
  gemm_wmma_f16<<<dim3(N_ / 128, QKV_ / 64), 256, 0, stream>>>(xm, WtQX, bqkv_x, qkv, N_, DX_, QKV_);
  gemm_wmma_f16<<<dim3(L_ / 128, QKV_ / 64), 256, 0, stream>>>(ym, WtQY, bqkv_y,
                                                               qkv + (size_t)N_ * QKV_, L_, DY_, QKV_);

  // 4) per-head QK norm + RoPE; build Q,K [H][T][D] and V^T [H][D][T]
  qknorm_rope_kernel<<<T_ * H_, 128, 0, stream>>>(qkv, rope_cos, rope_sin,
                                                  qnx, knx, qny, kny, Qb, Kb, Vt);

  // 5) flash attention
  attn_wmma_kernel<<<dim3(T_ / 128, H_), 256, 0, stream>>>(Qb, Kb, Vt, aout);

  // 6) output projections straight into d_out (x_out then y_out)
  float* xo = (float*)d_out;
  float* yo = xo + (size_t)N_ * DX_;
  gemm_wmma_f16<<<dim3(N_ / 128, DX_ / 64), 256, 0, stream>>>(aout, WtPX, bproj_x, xo, N_, DX_, DX_);
  gemm_wmma_f16<<<dim3(L_ / 128, DY_ / 64), 256, 0, stream>>>(aout + (size_t)N_ * DX_, WtPY, bproj_y,
                                                              yo, L_, DX_, DY_);
  (void)in_sizes; (void)n_in; (void)out_size; (void)ws_size;
}
